// LfExpert_6451040879172
// MI455X (gfx1250) — matrix operand order repaired and checked by reading, not compiled, not run on hardware
//
#include <hip/hip_runtime.h>
#include <hip/hip_bf16.h>

typedef __attribute__((ext_vector_type(16))) __bf16 v16bf;
typedef __attribute__((ext_vector_type(8)))  __bf16 v8bf;
typedef __attribute__((ext_vector_type(8)))  float  v8f;

#define LEAK_   0.2f
#define CLAMP_  0.8f
#define HW_     128
#define P_      16384
#define C_      64

__device__ __forceinline__ v16bf cat8(v8bf lo, v8bf hi) {
  return __builtin_shufflevector(lo, hi, 0, 1, 2, 3, 4, 5, 6, 7,
                                         8, 9, 10, 11, 12, 13, 14, 15);
}

// ---------------- weight f32 -> bf16 conversion ----------------
__global__ void cvt_bf16_k(const float* __restrict__ s, __bf16* __restrict__ d, int n) {
  int i = blockIdx.x * 256 + threadIdx.x;
  if (i < n) d[i] = (__bf16)s[i];
}

// ---------------- 3x3 conv via implicit-GEMM WMMA ----------------
// in/out: [B][64][P] f32. w: bf16 [64][576] (oc, ic*9+dy*3+dx).
// MODE 0: out = acc+bias (+res0 if R0)(+res1 if R1)
// MODE 1: out = leaky(acc+bias), per-batch skip when cof==0   (expert conv1)
// MODE 2: out += cof * (acc+bias), per-batch skip when cof==0 (expert conv2)
// Workgroup: 128 threads = 4 waves (oc tiles) x 32 pixels (2 N-tiles).
template <int MODE, bool R0, bool R1>
__global__ __launch_bounds__(128)
void conv3x3_t(const float* __restrict__ in, const __bf16* __restrict__ w,
               const float* __restrict__ bias, float* __restrict__ out,
               const float* __restrict__ res0, const float* __restrict__ res1,
               const float* __restrict__ cof, int expert) {
  const int tid  = threadIdx.x;
  const int wave = tid >> 5, lane = tid & 31;
  const int half = lane >> 4, ln = lane & 15;
  const int blk  = blockIdx.x;            // 4096 = B * HW * (HW/32)
  const int b    = blk >> 9;
  const int rem  = blk & 511;
  const int y    = rem >> 2;
  const int x0   = (rem & 3) << 5;

  float cf = 1.f;
  if (MODE != 0) { cf = cof[b * 4 + expert]; if (cf == 0.f) return; }

  // halo patch: [ic 0..63][ry 0..2][cx 0..33]
  __shared__ __bf16 patch[64 * 3 * 34];    // 6528
  // B panel in WMMA-fragment element order: [kk 0..17][nt 0..1][lane 0..31][j 0..15]
  __shared__ __bf16 panel[18 * 2 * 32 * 16]; // 18432

  const size_t bofs = (size_t)b * C_ * P_;
  const float* inb = in + bofs;
  for (int t = tid; t < 64 * 3 * 34; t += 128) {
    int ic = t / 102, r = t % 102, ry = r / 34, cx = r % 34;
    int yy = y + ry - 1, xx = x0 + cx - 1;
    float v = 0.f;
    if (yy >= 0 && yy < HW_ && xx >= 0 && xx < HW_) v = inb[(size_t)ic * P_ + yy * HW_ + xx];
    patch[t] = (__bf16)v;
  }
  __syncthreads();

  // im2col gather into fragment order (done once, shared by all 4 waves)
  for (int t = tid; t < 18 * 1024; t += 128) {
    int j  = t & 15;
    int l  = (t >> 4) & 31;
    int nt = (t >> 9) & 1;
    int kk = t >> 10;
    int k  = kk * 32 + ((l & 16) >> 1) + ((j & 8) << 1) + (j & 7); // fragment element j in table K order
    int n  = (l & 15) + (nt << 4);
    int ic = k / 9, t9 = k % 9;
    panel[t] = patch[(ic * 3 + t9 / 3) * 34 + n + (t9 % 3)];
  }
  __syncthreads();

  const int ocb = wave << 4;               // 4 waves -> 64 oc
  const __bf16* wrow = w + (size_t)(ocb + ln) * 576 + half * 8;
  const __bf16* prow = &panel[lane * 16];
  v8f acc0 = {}, acc1 = {};
  for (int kk = 0; kk < 18; ++kk) {        // K = 576 = 18 * 32
    v8bf alo = *(const v8bf*)(wrow + kk * 32);
    v8bf ahi = *(const v8bf*)(wrow + kk * 32 + 16);
    v16bf af = cat8(alo, ahi);
    const __bf16* pp = prow + kk * 1024;
    v16bf bf0 = cat8(*(const v8bf*)(pp),       *(const v8bf*)(pp + 8));
    v16bf bf1 = cat8(*(const v8bf*)(pp + 512), *(const v8bf*)(pp + 520));
    acc0 = __builtin_amdgcn_wmma_f32_16x16x32_bf16(false, af, false, bf0, (short)0, acc0, false, false);
    acc1 = __builtin_amdgcn_wmma_f32_16x16x32_bf16(false, af, false, bf1, (short)0, acc1, false, false);
  }

  float* ob = out + bofs;
  #pragma unroll
  for (int nt = 0; nt < 2; ++nt) {
    const int px = y * HW_ + x0 + (nt << 4) + ln;
    #pragma unroll
    for (int r = 0; r < 8; ++r) {
      int oc = ocb + r + 8 * half;
      size_t pos = (size_t)oc * P_ + px;
      float v = (nt ? acc1[r] : acc0[r]) + bias[oc];
      if (R0) v += res0[bofs + pos];
      if (R1) v += res1[bofs + pos];
      if (MODE == 1) v = (v > 0.f) ? v : LEAK_ * v;
      if (MODE == 2) ob[pos] += cf * v;
      else           ob[pos] = v;
    }
  }
}

// ---------------- 1x1 conv (pointwise GEMM) via WMMA ----------------
// input channels k<K0 from in0 (batch stride bs0), else in1. Output split
// oc<64 -> out0, oc>=64 -> out1 (each [B][64][P]).
__global__ __launch_bounds__(256)
void pw_wmma_k(const float* __restrict__ in0, const float* __restrict__ in1,
               int K0, size_t bs0, size_t bs1,
               const __bf16* __restrict__ w, const float* __restrict__ bias,
               float* __restrict__ out0, float* __restrict__ out1,
               int M, int K) {
  const int tid  = threadIdx.x;
  const int wave = tid >> 5, lane = tid & 31;
  const int half = lane >> 4, ln = lane & 15;
  const int b  = blockIdx.x >> 10;         // 8192 = B * (P/16)
  const int p0 = (blockIdx.x & 1023) << 4;

  __shared__ __bf16 stage[128 * 16];       // [k][n], coalesced staging
  __shared__ __bf16 panel[4 * 512];        // fragment order [kk][lane][j]
  for (int t = tid; t < K * 16; t += blockDim.x) {
    int k = t >> 4, n = t & 15;
    float v = (k < K0) ? in0[(size_t)b * bs0 + (size_t)k * P_ + p0 + n]
                       : in1[(size_t)b * bs1 + (size_t)(k - K0) * P_ + p0 + n];
    stage[t] = (__bf16)v;
  }
  __syncthreads();
  for (int t = tid; t < K * 16; t += blockDim.x) {
    int j = t & 15, l = (t >> 4) & 31, kk = t >> 9;
    int k = kk * 32 + ((l & 16) >> 1) + ((j & 8) << 1) + (j & 7);
    panel[t] = stage[k * 16 + (l & 15)];
  }
  __syncthreads();

  const int ocb = wave << 4;
  if (ocb >= M) return;
  const __bf16* wrow = w + (size_t)(ocb + ln) * K + half * 8;
  const __bf16* prow = &panel[lane * 16];
  v8f acc = {};
  for (int kk = 0; kk < (K >> 5); ++kk) {
    v16bf af = cat8(*(const v8bf*)(wrow + kk * 32), *(const v8bf*)(wrow + kk * 32 + 16));
    const __bf16* pp = prow + kk * 512;
    v16bf bf = cat8(*(const v8bf*)(pp), *(const v8bf*)(pp + 8));
    acc = __builtin_amdgcn_wmma_f32_16x16x32_bf16(false, af, false, bf, (short)0, acc, false, false);
  }
  const int px = p0 + ln;
  #pragma unroll
  for (int r = 0; r < 8; ++r) {
    int oc = ocb + r + 8 * half;
    float v = acc[r] + (bias ? bias[oc] : 0.f);
    if (oc < 64) out0[(size_t)b * C_ * P_ + (size_t)oc * P_ + px] = v;
    else         out1[(size_t)b * C_ * P_ + (size_t)(oc - 64) * P_ + px] = v;
  }
}

// ---------------- instance norm, in place on channels 0..31 ----------------
__global__ void inorm_k(float* __restrict__ buf, const float* __restrict__ g,
                        const float* __restrict__ bn) {
  int blk = blockIdx.x;                    // 256 = B * 32
  int b = blk >> 5, c = blk & 31;
  float* p = buf + ((size_t)b * C_ + c) * P_;
  __shared__ float s1[256], s2[256];
  float a = 0.f, q = 0.f;
  for (int i = threadIdx.x; i < P_; i += 256) { float v = p[i]; a += v; q += v * v; }
  s1[threadIdx.x] = a; s2[threadIdx.x] = q; __syncthreads();
  for (int st = 128; st > 0; st >>= 1) {
    if (threadIdx.x < st) { s1[threadIdx.x] += s1[threadIdx.x + st];
                            s2[threadIdx.x] += s2[threadIdx.x + st]; }
    __syncthreads();
  }
  float mean = s1[0] * (1.f / P_);
  float var  = s2[0] * (1.f / P_) - mean * mean;  // ddof=0 (jnp.var default)
  float inv  = rsqrtf(var + 1e-5f) * g[c];
  float beta = bn[c];
  for (int i = threadIdx.x; i < P_; i += 256) p[i] = (p[i] - mean) * inv + beta;
}

// ---------------- s = 0.8*(2*sigmoid(t)-1) = 0.8*tanh(t/2); tB = x2*exp(s) ----
__global__ void ewexp_k(float* __restrict__ tB, const float* __restrict__ x2, size_t n) {
  size_t i = (size_t)blockIdx.x * 256 + threadIdx.x;
  if (i >= n) return;
  float s = CLAMP_ * tanhf(0.5f * tB[i]);
  tB[i] = x2[i] * expf(s);
}

__global__ void copy_k(float* __restrict__ d, const float* __restrict__ s, size_t n) {
  size_t i = (size_t)blockIdx.x * 256 + threadIdx.x;
  if (i < n) d[i] = s[i];
}

// ---------------- global max+mean pooling: g[b][c] = max + mean ----------------
__global__ void pool_k(const float* __restrict__ xf, float* __restrict__ g) {
  int blk = blockIdx.x;                    // 512 = B * 64
  int b = blk >> 6, c = blk & 63;
  const float* p = xf + ((size_t)b * C_ + c) * P_;
  __shared__ float sm[256], sx[256];
  float a = 0.f, mx = -3.4e38f;
  for (int i = threadIdx.x; i < P_; i += 256) { float v = p[i]; a += v; mx = fmaxf(mx, v); }
  sm[threadIdx.x] = a; sx[threadIdx.x] = mx; __syncthreads();
  for (int st = 128; st > 0; st >>= 1) {
    if (threadIdx.x < st) { sm[threadIdx.x] += sm[threadIdx.x + st];
                            sx[threadIdx.x] = fmaxf(sx[threadIdx.x], sx[threadIdx.x + st]); }
    __syncthreads();
  }
  if (threadIdx.x == 0) g[b * C_ + c] = sx[0] + sm[0] * (1.f / P_);
}

// ---------------- MoE gate: noisy top-k (E=4, K=2) + masked softmax ----------
__global__ void gate_k(const float* __restrict__ g,
                       const float* __restrict__ gw0, const float* __restrict__ gb0,
                       const float* __restrict__ gw1, const float* __restrict__ gb1,
                       float* __restrict__ cof, float* __restrict__ cof_out) {
  int b = threadIdx.x;
  if (b >= 8) return;
  const float* gb = g + b * C_;
  float lg[4], nz[4];
  for (int e = 0; e < 4; ++e) {
    float a0 = gb0[e], a1 = gb1[e];
    for (int c = 0; c < C_; ++c) { float gv = gb[c]; a0 += gv * gw0[e * C_ + c]; a1 += gv * gw1[e * C_ + c]; }
    lg[e] = (a1 > 0.f) ? a1 : LEAK_ * a1;
    nz[e] = (a0 > 20.f) ? a0 : log1pf(expf(a0));  // softplus
  }
  float nm = 0.25f * (nz[0] + nz[1] + nz[2] + nz[3]);
  float ss = 0.f;
  for (int e = 0; e < 4; ++e) { float d = nz[e] - nm; ss += d * d; }
  float ns = sqrtf(ss / 3.f);                     // ddof=1
  float sc[4];
  for (int e = 0; e < 4; ++e) sc[e] = lg[e] + (nz[e] - nm) / ns;
  int i0 = 0, i1 = -1; float b0 = -3.4e38f, b1 = -3.4e38f;
  for (int e = 0; e < 4; ++e) {                   // stable top-2
    if (sc[e] > b0)      { b1 = b0; i1 = i0; b0 = sc[e]; i0 = e; }
    else if (sc[e] > b1) { b1 = sc[e]; i1 = e; }
  }
  float m  = fmaxf(lg[i0], lg[i1]);
  float e0 = expf(lg[i0] - m), e1 = expf(lg[i1] - m);
  float den = e0 + e1;
  float c4[4] = {0.f, 0.f, 0.f, 0.f};
  c4[i0] = e0 / den; c4[i1] = e1 / den;
  for (int e = 0; e < 4; ++e) { cof[b * 4 + e] = c4[e]; cof_out[b * 4 + e] = c4[e]; }
}

// ---------------- host-side orchestration ----------------
extern "C" void kernel_launch(void* const* d_in, const int* in_sizes, int n_in,
                              void* d_out, int out_size, void* d_ws, size_t ws_size,
                              hipStream_t stream) {
  (void)in_sizes; (void)n_in; (void)out_size; (void)ws_size;
  const float* x      = (const float*)d_in[0];
  const float* winv   = (const float*)d_in[1];
  const float* blk_w1 = (const float*)d_in[2];
  const float* blk_b1 = (const float*)d_in[3];
  const float* blk_nw = (const float*)d_in[4];
  const float* blk_nb = (const float*)d_in[5];
  const float* blk_w2 = (const float*)d_in[6];
  const float* blk_b2 = (const float*)d_in[7];
  const float* fuse_w = (const float*)d_in[8];
  const float* fuse_b = (const float*)d_in[9];
  const float* gw0    = (const float*)d_in[10];
  const float* gb0    = (const float*)d_in[11];
  const float* gw1    = (const float*)d_in[12];
  const float* gb1    = (const float*)d_in[13];
  const float* ew1    = (const float*)d_in[14];
  const float* eb1    = (const float*)d_in[15];
  const float* ew2    = (const float*)d_in[16];
  const float* eb2    = (const float*)d_in[17];

  const size_t S = (size_t)8 * C_ * P_;           // 8,388,608 elements / buffer
  float* ws   = (float*)d_ws;
  float* x1   = ws;
  float* x2   = x1 + S;
  float* tA   = x2 + S;
  float* tB   = tA + S;
  float* y1   = tB + S;
  float* y2   = y1 + S;
  float* xf   = y2 + S;
  float* gbuf = xf + S;                           // 512
  float* cofb = gbuf + 512;                       // 32 (+pad)
  __bf16* w_inv = (__bf16*)(cofb + 64);
  __bf16* w_b1  = w_inv + 16384;                  // 3 * 64 * 576
  __bf16* w_b2  = w_b1 + 110592;
  __bf16* w_f   = w_b2 + 110592;                  // 64 * 128
  __bf16* w_e1  = w_f + 8192;                     // 4 * 64 * 576
  __bf16* w_e2  = w_e1 + 147456;

  cvt_bf16_k<<<(16384 + 255) / 256, 256, 0, stream>>>(winv,   w_inv, 16384);
  cvt_bf16_k<<<(110592 + 255) / 256, 256, 0, stream>>>(blk_w1, w_b1, 110592);
  cvt_bf16_k<<<(110592 + 255) / 256, 256, 0, stream>>>(blk_w2, w_b2, 110592);
  cvt_bf16_k<<<(8192 + 255) / 256, 256, 0, stream>>>(fuse_w, w_f, 8192);
  cvt_bf16_k<<<(147456 + 255) / 256, 256, 0, stream>>>(ew1,   w_e1, 147456);
  cvt_bf16_k<<<(147456 + 255) / 256, 256, 0, stream>>>(ew2,   w_e2, 147456);

  const int CG = 4096;                            // B * HW * (HW/32)
  const int PG = 8192;                            // B * (P/16)
  const size_t EW = (S + 255) / 256;

  // z = winv @ x : oc<64 -> x1, oc>=64 -> x2
  pw_wmma_k<<<PG, 256, 0, stream>>>(x, nullptr, 128, (size_t)128 * P_, 0,
                                    w_inv, nullptr, x1, x2, 128, 128);

  // y1 = x1 + hin0(x2) = x1 + x2 + conv2(inorm|cat(conv1(x2)))
  conv3x3_t<0, false, false><<<CG, 128, 0, stream>>>(x2, w_b1, blk_b1, tA, nullptr, nullptr, nullptr, 0);
  inorm_k<<<256, 256, 0, stream>>>(tA, blk_nw, blk_nb);
  conv3x3_t<0, true, true><<<CG, 128, 0, stream>>>(tA, w_b2, blk_b2, y1, x2, x1, nullptr, 0);

  // tB = hin2(y1) = y1 + conv2(...)
  conv3x3_t<0, false, false><<<CG, 128, 0, stream>>>(y1, w_b1 + 2 * 36864, blk_b1 + 128, tA, nullptr, nullptr, nullptr, 0);
  inorm_k<<<256, 256, 0, stream>>>(tA, blk_nw + 64, blk_nb + 64);
  conv3x3_t<0, true, false><<<CG, 128, 0, stream>>>(tA, w_b2 + 2 * 36864, blk_b2 + 128, tB, y1, nullptr, nullptr, 0);

  // tB = x2 * exp(0.8 * (2*sigmoid(tB) - 1))
  ewexp_k<<<EW, 256, 0, stream>>>(tB, x2, S);

  // y2 = x2*exp(s) + hin1(y1) = conv2(...) + y1 + tB
  conv3x3_t<0, false, false><<<CG, 128, 0, stream>>>(y1, w_b1 + 36864, blk_b1 + 64, tA, nullptr, nullptr, nullptr, 0);
  inorm_k<<<256, 256, 0, stream>>>(tA, blk_nw + 32, blk_nb + 32);
  conv3x3_t<0, true, true><<<CG, 128, 0, stream>>>(tA, w_b2 + 36864, blk_b2 + 64, y2, y1, tB, nullptr, 0);

  // xf = fuse_w @ cat(y1, y2) + fuse_b
  pw_wmma_k<<<PG, 128, 0, stream>>>(y1, y2, 64, (size_t)64 * P_, (size_t)64 * P_,
                                    w_f, fuse_b, xf, nullptr, 64, 128);

  // gate
  pool_k<<<512, 256, 0, stream>>>(xf, gbuf);
  gate_k<<<1, 32, 0, stream>>>(gbuf, gw0, gb0, gw1, gb1, cofb, (float*)d_out + S);

  // out = xf + sum_e cof_e * conv2_e(leaky(conv1_e(xf)))   (sum cof == 1)
  copy_k<<<EW, 256, 0, stream>>>((float*)d_out, xf, S);
  for (int e = 0; e < 4; ++e) {
    conv3x3_t<1, false, false><<<CG, 128, 0, stream>>>(xf, w_e1 + e * 36864, eb1 + e * 64, tA,
                                                       nullptr, nullptr, cofb, e);
    conv3x3_t<2, false, false><<<CG, 128, 0, stream>>>(tA, w_e2 + e * 36864, eb2 + e * 64, (float*)d_out,
                                                       nullptr, nullptr, cofb, e);
  }
}
